// EncoderLayer_61907658605192
// MI455X (gfx1250) — matrix operand order repaired and checked by reading, not compiled, not run on hardware
//
#include <hip/hip_runtime.h>
#include <math.h>

// ---------------------------------------------------------------------------
// Types for CDNA5 WMMA (wave32): v_wmma_f32_16x16x32_bf16
// ---------------------------------------------------------------------------
typedef __bf16 bf16;
typedef __attribute__((ext_vector_type(8)))  float v8f;
typedef __attribute__((ext_vector_type(16))) bf16  v16bf;
typedef __attribute__((ext_vector_type(8)))  bf16  v8bf;

union Frag { v16bf v; v8bf h[2]; };

// Problem constants (match reference)
#define BQ  2
#define SQ  1024
#define DQ  1024
#define HQ  16
#define HDQ 64
#define EQ  8
#define FFQ 4096

// GEMM tiling
#define BM   128
#define BN   128
#define BKT  32
#define LDST 40     // padded LDS row stride (elements) -> 80B rows, 16B aligned chunks

// Epilogue modes
#define EPI_F32_BIAS  0   // Cf = acc + bias[col]
#define EPI_SCALE_F32 1   // Cf = acc * alpha
#define EPI_BF16      2   // Cb = (bf16)acc
#define EPI_GELU_BF16 3   // Cb = (bf16)gelu(acc + bias[col])   (exact erf gelu)
#define EPI_MOE_ACC   4   // Cf += rowscale[row*rs_stride] * (acc + bias[col])

// ---------------------------------------------------------------------------
// CDNA5 async global->LDS copy (ASYNCcnt-tracked DMA, no VGPR round trip)
// ---------------------------------------------------------------------------
__device__ __forceinline__ unsigned lds_addr_of(const void* p)
{
    // generic -> addrspace(3) -> 32-bit LDS byte offset
    return (unsigned)(size_t)(const __attribute__((address_space(3))) void*)p;
}

__device__ __forceinline__ void async_ld_b128(unsigned lds, const void* g)
{
    asm volatile("global_load_async_to_lds_b128 %0, %1, off"
                 :: "v"(lds), "v"(g) : "memory");
}

__device__ __forceinline__ void wait_async0()
{
    asm volatile("s_wait_asynccnt 0" ::: "memory");
}

// ---------------------------------------------------------------------------
// bf16 WMMA GEMM:  C[M,N] = A[M,K] * B^T  (+epilogue)
//   A is [M,K] row-major, B is [N,K] row-major (K contiguous for both).
//   GUARD=false: M,N multiples of 128 -> branch-free async-DMA staging.
//   GUARD=true : bounds-checked synchronous staging (used for P*V, N=64).
// ---------------------------------------------------------------------------
template<bool GUARD>
__global__ __launch_bounds__(256, 1)
void gemm_bf16_kernel(const bf16* __restrict__ A, int lda,
                      const bf16* __restrict__ Bm, int ldb,
                      float* __restrict__ Cf, bf16* __restrict__ Cb, int ldc,
                      const float* __restrict__ bias,
                      const float* __restrict__ rowscale, int rs_stride,
                      int M, int N, int Kd, float alpha, int mode)
{
    __shared__ __align__(16) bf16 As[2][BM * LDST];
    __shared__ __align__(16) bf16 Bs[2][BN * LDST];

    const int t    = threadIdx.x;
    const int m0   = blockIdx.y * BM;
    const int n0   = blockIdx.x * BN;
    const int lane = t & 31;
    const int wave = t >> 5;
    const int wm   = (wave >> 2) * 64;   // wave row offset within block tile
    const int wn   = (wave & 3)  * 32;   // wave col offset within block tile
    const int fr   = lane & 15;
    const int hi   = lane >> 4;          // lane half (ISA 16-bit A/B layout)

    v8f acc[4][2];
#pragma unroll
    for (int i = 0; i < 4; ++i)
#pragma unroll
        for (int j = 0; j < 2; ++j)
#pragma unroll
            for (int r = 0; r < 8; ++r) acc[i][j][r] = 0.0f;

    const int nk = Kd / BKT;

    auto stage = [&](int buf, int kt) {
        const int k0 = kt * BKT;
        if (!GUARD) {
            // Branch-free: both tiles are contiguous 16B row chunks -> async DMA.
#pragma unroll
            for (int c = 0; c < 2; ++c) {
                int lin = t * 16 + c * 8;          // element idx in 128x32 tile
                int row = lin >> 5, col = lin & 31;
                async_ld_b128(lds_addr_of(&As[buf][row * LDST + col]),
                              A + (size_t)(m0 + row) * lda + (k0 + col));
                async_ld_b128(lds_addr_of(&Bs[buf][row * LDST + col]),
                              Bm + (size_t)(n0 + row) * ldb + (k0 + col));
            }
        } else {
#pragma unroll
            for (int c = 0; c < 2; ++c) {
                int lin = t * 16 + c * 8;
                int row = lin >> 5, col = lin & 31;
                v8bf va, vb;
#pragma unroll
                for (int q = 0; q < 8; ++q) { va[q] = (bf16)0.0f; vb[q] = (bf16)0.0f; }
                if (m0 + row < M)
                    va = *(const v8bf*)(A + (size_t)(m0 + row) * lda + (k0 + col));
                if (n0 + row < N)
                    vb = *(const v8bf*)(Bm + (size_t)(n0 + row) * ldb + (k0 + col));
                *(v8bf*)(&As[buf][row * LDST + col]) = va;
                *(v8bf*)(&Bs[buf][row * LDST + col]) = vb;
            }
        }
    };
    auto stage_fence = [&]() { if (!GUARD) wait_async0(); };

    auto compute = [&](int buf) {
        Frag af[4], bfr[2];
        // ISA 16-bit A 16x32 layout: lane<16 -> K {0..7,16..23}; lane>=16 -> {8..15,24..31}
#pragma unroll
        for (int i = 0; i < 4; ++i) {
            const bf16* ap = &As[buf][(wm + i * 16 + fr) * LDST + hi * 8];
            af[i].h[0] = *(const v8bf*)ap;
            af[i].h[1] = *(const v8bf*)(ap + 16);
        }
        // B 32x16 layout: lane<16 -> column fr, K {0..7,16..23}; lane>=16 -> K {8..15,24..31}
#pragma unroll
        for (int j = 0; j < 2; ++j) {
            const bf16* bp = &Bs[buf][(wn + j * 16 + fr) * LDST + hi * 8];
            bfr[j].h[0] = *(const v8bf*)bp;
            bfr[j].h[1] = *(const v8bf*)(bp + 16);
        }
#pragma unroll
        for (int i = 0; i < 4; ++i)
#pragma unroll
            for (int j = 0; j < 2; ++j)
                acc[i][j] = __builtin_amdgcn_wmma_f32_16x16x32_bf16(
                    false, af[i].v, false, bfr[j].v, (short)0, acc[i][j],
                    false, false);
    };

    stage(0, 0);
    stage_fence();
    __syncthreads();
    int cur = 0;
    for (int kt = 0; kt < nk; ++kt) {
        if (kt + 1 < nk) stage(cur ^ 1, kt + 1);
        if (kt + 2 < nk) {
            // speculative L2 prefetch of the tile after next -> global_prefetch_b8
            int lin = t * 16;
            __builtin_prefetch(A + (size_t)(m0 + (lin >> 5)) * lda
                                 + (kt + 2) * BKT + (lin & 31), 0, 1);
            __builtin_prefetch(Bm + (size_t)(n0 + (lin >> 5)) * ldb
                                 + (kt + 2) * BKT + (lin & 31), 0, 1);
        }
        compute(cur);
        stage_fence();            // all async DMA into buf^1 landed
        __syncthreads();
        cur ^= 1;
    }

    // ---- Epilogue: C/D VGPR r, lanes0-15 -> (M=r,N=lane); lanes16-31 -> (M=r+8)
#pragma unroll
    for (int i = 0; i < 4; ++i) {
#pragma unroll
        for (int j = 0; j < 2; ++j) {
#pragma unroll
            for (int r = 0; r < 8; ++r) {
                int row = m0 + wm + i * 16 + hi * 8 + r;
                int col = n0 + wn + j * 16 + fr;
                if (row < M && col < N) {
                    float v = acc[i][j][r];
                    size_t idx = (size_t)row * ldc + col;
                    float bb = bias ? bias[col] : 0.0f;
                    switch (mode) {
                    case EPI_F32_BIAS:  Cf[idx] = v + bb; break;
                    case EPI_SCALE_F32: Cf[idx] = v * alpha; break;
                    case EPI_BF16:      Cb[idx] = (bf16)v; break;
                    case EPI_GELU_BF16: {
                        float u = v + bb;
                        Cb[idx] = (bf16)(0.5f * u * (1.0f + erff(u * 0.70710678118f)));
                    } break;
                    case EPI_MOE_ACC: {
                        float w = rowscale[(size_t)row * rs_stride];
                        Cf[idx] += w * (v + bb);
                    } break;
                    }
                }
            }
        }
    }
}

// ---------------------------------------------------------------------------
// Elementwise f32 -> bf16 cast, and zero-fill
// ---------------------------------------------------------------------------
__global__ void cast_f32_bf16_kernel(const float* __restrict__ in,
                                     bf16* __restrict__ outp, int n)
{
    int i = blockIdx.x * blockDim.x + threadIdx.x;
    if (i < n) outp[i] = (bf16)in[i];
}

__global__ void zero_f32_kernel(float* __restrict__ p, int n)
{
    int i = blockIdx.x * blockDim.x + threadIdx.x;
    if (i < n) p[i] = 0.0f;
}

// ---------------------------------------------------------------------------
// Tiled transpose + cast: out[N,K] bf16 = (in[K,N] f32)^T, batched on blockIdx.z
// K,N multiples of 32. 32x8 threads, 32x32 tile via LDS.
// ---------------------------------------------------------------------------
__global__ __launch_bounds__(256)
void transpose_cast_kernel(const float* __restrict__ in, bf16* __restrict__ outp,
                           int K, int N)
{
    __shared__ float tile[32][33];
    const int tx = threadIdx.x & 31, ty = threadIdx.x >> 5;
    const int bx = blockIdx.x * 32;   // n origin
    const int by = blockIdx.y * 32;   // k origin
    const float* src = in + (size_t)blockIdx.z * K * N;
    bf16* dst = outp + (size_t)blockIdx.z * K * N;
#pragma unroll
    for (int i = 0; i < 32; i += 8)
        tile[ty + i][tx] = src[(size_t)(by + ty + i) * N + (bx + tx)];
    __syncthreads();
#pragma unroll
    for (int i = 0; i < 32; i += 8)
        dst[(size_t)(bx + ty + i) * K + (by + tx)] = (bf16)tile[tx][ty + i];
}

// ---------------------------------------------------------------------------
// RoPE + head pack: [B,S,D] f32 -> [B,H,S,HD] bf16 (Q,K with rope)
//                   or [B,H,HD,S] bf16 transposed  (V, no rope)
// one thread per (b,s,h,d2), d2 in [0,32): handles pair (d2, d2+32)
// ---------------------------------------------------------------------------
__global__ __launch_bounds__(256)
void rope_pack_kernel(const float* __restrict__ Qf, bf16* __restrict__ Qp,
                      int applyRope, int vtrans)
{
    int idx = blockIdx.x * blockDim.x + threadIdx.x;  // < B*S*H*32 = 2^20
    int d2 = idx & 31;
    int h  = (idx >> 5) & (HQ - 1);
    int s  = (idx >> 9) & (SQ - 1);
    int b  = idx >> 19;
    const float* src = Qf + ((size_t)(b * SQ + s) * DQ + h * HDQ);
    float q0 = src[d2], q1 = src[d2 + 32];
    float o0, o1;
    if (applyRope) {
        // inv_freq = 10000^(-d2/32)
        float invf = __expf(-(float)d2 * (logf(10000.0f) / 32.0f));
        float ang = (float)s * invf;
        float c = __cosf(ang), sn = __sinf(ang);
        o0 = q0 * c - q1 * sn;       // q*cos + rotate_half(q)*sin, first half
        o1 = q1 * c + q0 * sn;       // second half
    } else { o0 = q0; o1 = q1; }
    if (vtrans) {
        bf16* dst = Qp + (size_t)(b * HQ + h) * HDQ * SQ;
        dst[(size_t)d2 * SQ + s]        = (bf16)o0;
        dst[(size_t)(d2 + 32) * SQ + s] = (bf16)o1;
    } else {
        bf16* dst = Qp + ((size_t)((b * HQ + h) * SQ + s) * HDQ);
        dst[d2]      = (bf16)o0;
        dst[d2 + 32] = (bf16)o1;
    }
}

// ---------------------------------------------------------------------------
// Row softmax (f32 in, bf16 probs out). One 256-thread block per row.
// ---------------------------------------------------------------------------
__global__ __launch_bounds__(256)
void softmax_rows_kernel(const float* __restrict__ S, bf16* __restrict__ P,
                         int ncols)
{
    __shared__ float red[256];
    const int t = threadIdx.x;
    const size_t base = (size_t)blockIdx.x * ncols;
    float mx = -3.4e38f;
    for (int c = t; c < ncols; c += 256) mx = fmaxf(mx, S[base + c]);
    red[t] = mx; __syncthreads();
    for (int s = 128; s > 0; s >>= 1) {
        if (t < s) red[t] = fmaxf(red[t], red[t + s]);
        __syncthreads();
    }
    mx = red[0]; __syncthreads();
    float sum = 0.0f;
    for (int c = t; c < ncols; c += 256) sum += __expf(S[base + c] - mx);
    red[t] = sum; __syncthreads();
    for (int s = 128; s > 0; s >>= 1) {
        if (t < s) red[t] += red[t + s];
        __syncthreads();
    }
    float inv = 1.0f / red[0];
    for (int c = t; c < ncols; c += 256)
        P[base + c] = (bf16)(__expf(S[base + c] - mx) * inv);
}

// ---------------------------------------------------------------------------
// Residual + LayerNorm over D=1024. One block per token row.
// ---------------------------------------------------------------------------
__global__ __launch_bounds__(256)
void residual_ln_kernel(const float* __restrict__ x, const float* __restrict__ y,
                        const float* __restrict__ g, const float* __restrict__ be,
                        float* __restrict__ outf, bf16* __restrict__ outb)
{
    __shared__ float red[256];
    __shared__ float mvar[2];
    const int t = threadIdx.x;
    const size_t base = (size_t)blockIdx.x * DQ;
    float v[4];
    float s = 0.0f, s2 = 0.0f;
#pragma unroll
    for (int i = 0; i < 4; ++i) {
        int c = t + i * 256;
        v[i] = x[base + c] + y[base + c];
        s += v[i]; s2 += v[i] * v[i];
    }
    red[t] = s; __syncthreads();
    for (int k = 128; k > 0; k >>= 1) { if (t < k) red[t] += red[t + k]; __syncthreads(); }
    if (t == 0) mvar[0] = red[0] * (1.0f / DQ);
    __syncthreads();
    red[t] = s2; __syncthreads();
    for (int k = 128; k > 0; k >>= 1) { if (t < k) red[t] += red[t + k]; __syncthreads(); }
    if (t == 0) {
        float mean = mvar[0];
        mvar[1] = red[0] * (1.0f / DQ) - mean * mean;  // population variance
    }
    __syncthreads();
    float mean = mvar[0];
    float inv = rsqrtf(mvar[1] + 1e-5f);
#pragma unroll
    for (int i = 0; i < 4; ++i) {
        int c = t + i * 256;
        float o = (v[i] - mean) * inv * g[c] + be[c];
        outf[base + c] = o;
        if (outb) outb[base + c] = (bf16)o;
    }
}

// ---------------------------------------------------------------------------
// Router: gate = softmax(x1 @ Wg + bg); top-2 renormalized -> comb[tok, E]
// ---------------------------------------------------------------------------
__global__ __launch_bounds__(256)
void router_kernel(const float* __restrict__ x1, const float* __restrict__ Wg,
                   const float* __restrict__ bg, float* __restrict__ comb)
{
    __shared__ float part[EQ][32];
    __shared__ float logit[EQ];
    const int t = threadIdx.x;
    const int e = t >> 5, lane = t & 31;
    const size_t tok = blockIdx.x;
    const float* xr = x1 + tok * DQ;
    float p = 0.0f;
    for (int d = lane; d < DQ; d += 32) p += xr[d] * Wg[(size_t)d * EQ + e];
    part[e][lane] = p;
    __syncthreads();
    if (t < EQ) {
        float s = bg[t];
        for (int i = 0; i < 32; ++i) s += part[t][i];
        logit[t] = s;
    }
    __syncthreads();
    if (t == 0) {
        float mx = logit[0];
        for (int i = 1; i < EQ; ++i) mx = fmaxf(mx, logit[i]);
        float probs[EQ]; float sum = 0.0f;
        for (int i = 0; i < EQ; ++i) { probs[i] = __expf(logit[i] - mx); sum += probs[i]; }
        float invs = 1.0f / sum;
        for (int i = 0; i < EQ; ++i) probs[i] *= invs;
        int i0 = 0;
        for (int i = 1; i < EQ; ++i) if (probs[i] > probs[i0]) i0 = i;
        int i1 = (i0 == 0) ? 1 : 0;
        for (int i = 0; i < EQ; ++i) if (i != i0 && probs[i] > probs[i1]) i1 = i;
        float ps = probs[i0] + probs[i1];
        for (int i = 0; i < EQ; ++i) comb[tok * EQ + i] = 0.0f;
        comb[tok * EQ + i0] = probs[i0] / ps;
        comb[tok * EQ + i1] = probs[i1] / ps;
    }
}

// ---------------------------------------------------------------------------
// Orchestration
// ---------------------------------------------------------------------------
extern "C" void kernel_launch(void* const* d_in, const int* in_sizes, int n_in,
                              void* d_out, int out_size, void* d_ws, size_t ws_size,
                              hipStream_t stream)
{
    const float* x   = (const float*)d_in[0];
    const float* Wq  = (const float*)d_in[1];  const float* bq  = (const float*)d_in[2];
    const float* Wk  = (const float*)d_in[3];  const float* bk  = (const float*)d_in[4];
    const float* Wv  = (const float*)d_in[5];  const float* bv  = (const float*)d_in[6];
    const float* Wo  = (const float*)d_in[7];  const float* bo  = (const float*)d_in[8];
    const float* Wg  = (const float*)d_in[9];  const float* bg  = (const float*)d_in[10];
    const float* W1  = (const float*)d_in[11]; const float* b1  = (const float*)d_in[12];
    const float* W2  = (const float*)d_in[13]; const float* b2  = (const float*)d_in[14];
    const float* g1  = (const float*)d_in[15]; const float* be1 = (const float*)d_in[16];
    const float* g2  = (const float*)d_in[17]; const float* be2 = (const float*)d_in[18];
    float* outp = (float*)d_out;

    const size_t MT = (size_t)BQ * SQ;   // 2048 tokens

    size_t off = 0;
    auto alloc = [&](size_t bytes) -> void* {
        void* p = (char*)d_ws + off;
        off += (bytes + 255) & ~(size_t)255;
        return p;
    };
    bf16*  xb     = (bf16*) alloc(MT * DQ * 2);
    bf16*  Wqt    = (bf16*) alloc((size_t)DQ * DQ * 2);        // [N,K] = W^T
    bf16*  Wkt    = (bf16*) alloc((size_t)DQ * DQ * 2);
    bf16*  Wvt    = (bf16*) alloc((size_t)DQ * DQ * 2);
    bf16*  Wot    = (bf16*) alloc((size_t)DQ * DQ * 2);
    bf16*  W1t    = (bf16*) alloc((size_t)EQ * DQ * FFQ * 2);  // per-e [FF,D]
    bf16*  W2t    = (bf16*) alloc((size_t)EQ * FFQ * DQ * 2);  // per-e [D,FF]
    float* Qf     = (float*)alloc(MT * DQ * 4);
    float* Kf     = (float*)alloc(MT * DQ * 4);
    float* Vf     = (float*)alloc(MT * DQ * 4);
    bf16*  Qp     = (bf16*) alloc(MT * DQ * 2);                // [B,H,S,HD]
    bf16*  Kp     = (bf16*) alloc(MT * DQ * 2);                // [B,H,S,HD]
    bf16*  Vt     = (bf16*) alloc(MT * DQ * 2);                // [B,H,HD,S]
    float* scores = (float*)alloc((size_t)SQ * SQ * 4);
    bf16*  Pb     = (bf16*) alloc((size_t)SQ * SQ * 2);
    bf16*  aob    = (bf16*) alloc(MT * DQ * 2);
    float* projf  = (float*)alloc(MT * DQ * 4);
    float* x1f    = (float*)alloc(MT * DQ * 4);
    bf16*  x1b    = (bf16*) alloc(MT * DQ * 2);
    float* comb   = (float*)alloc(MT * EQ * 4);
    bf16*  hb     = (bf16*) alloc(MT * FFQ * 2);
    float* moe    = (float*)alloc(MT * DQ * 4);

    auto gemmU = [&](const bf16* A, int lda, const bf16* Bm, int ldb,
                     float* Cf, bf16* Cb, int ldc, const float* bias,
                     const float* rs, int rss, int M, int N, int K,
                     float alpha, int mode) {
        dim3 g(N / BN, M / BM);
        gemm_bf16_kernel<false><<<g, 256, 0, stream>>>(
            A, lda, Bm, ldb, Cf, Cb, ldc, bias, rs, rss, M, N, K, alpha, mode);
    };
    auto gemmG = [&](const bf16* A, int lda, const bf16* Bm, int ldb,
                     float* Cf, bf16* Cb, int ldc,
                     int M, int N, int K, float alpha, int mode) {
        dim3 g((N + BN - 1) / BN, (M + BM - 1) / BM);
        gemm_bf16_kernel<true><<<g, 256, 0, stream>>>(
            A, lda, Bm, ldb, Cf, Cb, ldc, nullptr, nullptr, 0, M, N, K, alpha, mode);
    };
    auto transpose = [&](const float* src, bf16* dst, int K, int N, int batch) {
        dim3 g(N / 32, K / 32, batch);
        transpose_cast_kernel<<<g, 256, 0, stream>>>(src, dst, K, N);
    };

    // 1) One-time precision/layout pass: x -> bf16; all weights -> transposed bf16
    cast_f32_bf16_kernel<<<(int)(MT * DQ) / 256, 256, 0, stream>>>(x, xb, (int)(MT * DQ));
    transpose(Wq, Wqt, DQ, DQ, 1);
    transpose(Wk, Wkt, DQ, DQ, 1);
    transpose(Wv, Wvt, DQ, DQ, 1);
    transpose(Wo, Wot, DQ, DQ, 1);
    transpose(W1, W1t, DQ, FFQ, EQ);
    transpose(W2, W2t, FFQ, DQ, EQ);

    // 2) QKV projections: [2048,1024] = xb @ W + b
    gemmU(xb, DQ, Wqt, DQ, Qf, nullptr, DQ, bq, nullptr, 0,
          (int)MT, DQ, DQ, 1.0f, EPI_F32_BIAS);
    gemmU(xb, DQ, Wkt, DQ, Kf, nullptr, DQ, bk, nullptr, 0,
          (int)MT, DQ, DQ, 1.0f, EPI_F32_BIAS);
    gemmU(xb, DQ, Wvt, DQ, Vf, nullptr, DQ, bv, nullptr, 0,
          (int)MT, DQ, DQ, 1.0f, EPI_F32_BIAS);

    // 3) RoPE + pack (Q,K: [B,H,S,HD]; V: transposed [B,H,HD,S])
    {
        int total = BQ * SQ * HQ * 32;
        rope_pack_kernel<<<total / 256, 256, 0, stream>>>(Qf, Qp, 1, 0);
        rope_pack_kernel<<<total / 256, 256, 0, stream>>>(Kf, Kp, 1, 0);
        rope_pack_kernel<<<total / 256, 256, 0, stream>>>(Vf, Vt, 0, 1);
    }

    // 4) Attention per (b,h): scores = Q K^T / 8 ; softmax ; O = P V
    for (int bh = 0; bh < BQ * HQ; ++bh) {
        const bf16* Qh = Qp + (size_t)bh * SQ * HDQ;
        const bf16* Kh = Kp + (size_t)bh * SQ * HDQ;
        const bf16* Vh = Vt + (size_t)bh * HDQ * SQ;
        int b = bh / HQ, h = bh % HQ;
        // scores[S,S] f32 = (Q @ K^T) * (1/sqrt(64));  B = K is already [N,K]=[S,HD]
        gemmU(Qh, HDQ, Kh, HDQ, scores, nullptr, SQ, nullptr, nullptr, 0,
              SQ, SQ, HDQ, 0.125f, EPI_SCALE_F32);
        softmax_rows_kernel<<<SQ, 256, 0, stream>>>(scores, Pb, SQ);
        // O[S,64] bf16 = P @ V ; B = V^T is [N,K]=[64,S]; N=64 -> guarded kernel
        bf16* Ob = aob + (size_t)b * SQ * DQ + (size_t)h * HDQ;
        gemmG(Pb, SQ, Vh, SQ, nullptr, Ob, DQ, SQ, HDQ, SQ, 1.0f, EPI_BF16);
    }

    // 5) Output projection + residual + LN1
    gemmU(aob, DQ, Wot, DQ, projf, nullptr, DQ, bo, nullptr, 0,
          (int)MT, DQ, DQ, 1.0f, EPI_F32_BIAS);
    residual_ln_kernel<<<(int)MT, 256, 0, stream>>>(x, projf, g1, be1, x1f, x1b);

    // 6) Router: top-2 combine weights (== reference's dense comb vector)
    router_kernel<<<(int)MT, 256, 0, stream>>>(x1f, Wg, bg, comb);

    // 7) MoE: moe = sum_e comb[:,e] * (gelu(x1 @ W1[e] + b1[e]) @ W2[e] + b2[e])
    zero_f32_kernel<<<(int)(MT * DQ) / 256, 256, 0, stream>>>(moe, (int)(MT * DQ));
    for (int e = 0; e < EQ; ++e) {
        const bf16* W1e = W1t + (size_t)e * DQ * FFQ;   // [FF, D]
        const bf16* W2e = W2t + (size_t)e * FFQ * DQ;   // [D, FF]
        gemmU(x1b, DQ, W1e, DQ, nullptr, hb, FFQ, b1 + (size_t)e * FFQ,
              nullptr, 0, (int)MT, FFQ, DQ, 1.0f, EPI_GELU_BF16);
        gemmU(hb, FFQ, W2e, FFQ, moe, nullptr, DQ, b2 + (size_t)e * DQ,
              comb + e, EQ, (int)MT, DQ, FFQ, 1.0f, EPI_MOE_ACC);
    }

    // 8) Residual + LN2 -> final output (f32)
    residual_ln_kernel<<<(int)MT, 256, 0, stream>>>(x1f, moe, g2, be2, outp, nullptr);
}
